// ADSREncoderV3_22230750724662
// MI455X (gfx1250) — matrix operand order repaired and checked by reading, not compiled, not run on hardware
//
#include <hip/hip_runtime.h>
#include <hip/hip_bf16.h>
#include <math.h>

// ---------------------------------------------------------------------------
// ADSREncoderV3 for MI455X (gfx1250, wave32).
//   wav(32,1,262144) -> RMS(hop 512) -> log -> conv1d(1->32,k3)+GELU
//   -> 4x [depthwise k3 (dil 1/2/4/8) -> pointwise GEMM (WMMA f16->f32)
//          -> global BN (stats over B,P) -> SiLU]
//   -> onset-segment weighted prototype (32,64,512)
// ---------------------------------------------------------------------------

typedef __attribute__((ext_vector_type(16))) _Float16 v16h;
typedef __attribute__((ext_vector_type(8)))  _Float16 v8h;
typedef __attribute__((ext_vector_type(8)))  float    v8f;

union V16 { v16h v; v8h h[2]; };

#define BB   32
#define PP   512
#define HOPS 512
#define NMAX 16

// ---------------- RMS: one wave32 per 512-sample window --------------------
__global__ void rms_kernel(const float* __restrict__ wav, float* __restrict__ lr, int nwin)
{
    int wave = (blockIdx.x * blockDim.x + threadIdx.x) >> 5;
    int lane = threadIdx.x & 31;
    if (wave >= nwin) return;
    const float4* base = (const float4*)(wav + (size_t)wave * HOPS + lane * 16);
    float s = 0.f;
#pragma unroll
    for (int j = 0; j < 4; ++j) {
        float4 v = base[j];
        s += v.x * v.x + v.y * v.y + v.z * v.z + v.w * v.w;
    }
#pragma unroll
    for (int m = 16; m >= 1; m >>= 1) s += __shfl_xor(s, m, 32);
    if (lane == 0) {
        float rms = sqrtf(s * (1.0f / HOPS) + 1e-7f);
        lr[wave] = logf(rms + 1e-7f);
    }
}

// ---------------- conv0 (1->32, k=3, pad 1) + exact GELU -------------------
__global__ void conv0_kernel(const float* __restrict__ lr, const float* __restrict__ w0,
                             const float* __restrict__ b0, float* __restrict__ x0)
{
    int idx = blockIdx.x * blockDim.x + threadIdx.x;
    if (idx >= BB * 32 * PP) return;
    int p = idx & 511;
    int c = (idx >> 9) & 31;
    int b = idx >> 14;
    const float* l = lr + b * PP;
    float xm = (p > 0)   ? l[p - 1] : 0.f;
    float xc = l[p];
    float xp = (p < 511) ? l[p + 1] : 0.f;
    float v = w0[c * 3 + 0] * xm + w0[c * 3 + 1] * xc + w0[c * 3 + 2] * xp + b0[c];
    x0[idx] = 0.5f * v * (1.0f + erff(v * 0.70710678118654752f));
}

// ------- depthwise(k3,dil) + pointwise GEMM via WMMA + BN-stat atomics -----
// One workgroup (128 thr = 4 waves) per (batch, 64-position tile).
// Pointwise: Y(64 x 64pos) = PW(64 x CI) * Z(CI x 64pos), f16 in / f32 acc.
// Z is stored position-major in LDS so every lane's B-fragment is a
// pair of 16B runs (2x ds_load_b128); A-fragments likewise.
template <int CI>
__global__ void __launch_bounds__(128)
dwpw_kernel(const float* __restrict__ x, const float* __restrict__ dw,
            const float* __restrict__ pw, float* __restrict__ y,
            float* __restrict__ ssum, float* __restrict__ ssq, int dil)
{
    __shared__ __align__(16) float    xs[CI][80];  // halo tile (pad 8/side; max dil 8)
    __shared__ __align__(16) _Float16 ztr[64][CI]; // depthwise out, position-major
    __shared__ __align__(16) _Float16 ah[64][CI];  // pointwise weights (A operand)

    const int tid = threadIdx.x;
    const int b   = blockIdx.x >> 3;
    const int p0  = (blockIdx.x & 7) * 64;

    for (int idx = tid; idx < CI * 80; idx += 128) {
        int i = idx / 80, pl = idx - i * 80;
        int gp = p0 + pl - 8;
        xs[i][pl] = (gp >= 0 && gp < PP) ? x[((size_t)b * CI + i) * PP + gp] : 0.f;
    }
    for (int idx = tid; idx < 64 * CI; idx += 128)
        ah[idx / CI][idx % CI] = (_Float16)pw[idx];
    __syncthreads();

    // depthwise conv, written transposed: ztr[pos][chan]
    for (int idx = tid; idx < CI * 64; idx += 128) {
        int i = idx & (CI - 1), n = idx / CI;
        int c0 = n + 8;
        float z = dw[i * 3 + 0] * xs[i][c0 - dil] +
                  dw[i * 3 + 1] * xs[i][c0] +
                  dw[i * 3 + 2] * xs[i][c0 + dil];
        ztr[n][i] = (_Float16)z;
    }
    __syncthreads();

    // Wave w owns output row-tile mt=w; loop over 4 column tiles.
    const int wv   = tid >> 5;
    const int lane = tid & 31;
    const int half = lane >> 4;   // lane-half per ISA wave32 fragment layout
    const int l15  = lane & 15;

    // A fragments depend only on kk: hoist out of the nt loop.
    // A (16x32 f16): M = l15, K(j) = (j>>3)*16 + half*8 + (j&7)
    V16 afrag[CI / 32];
#pragma unroll
    for (int kk = 0; kk < CI / 32; ++kk) {
        const v8h* arow = (const v8h*)&ah[wv * 16 + l15][kk * 32];
        afrag[kk].h[0] = arow[half];      // K = half*8 + 0..7
        afrag[kk].h[1] = arow[2 + half];  // K = 16 + half*8 + 0..7
    }

    float sacc[8], qacc[8];
#pragma unroll
    for (int r = 0; r < 8; ++r) { sacc[r] = 0.f; qacc[r] = 0.f; }

#pragma unroll
    for (int nt = 0; nt < 4; ++nt) {
        v8f acc = {};
#pragma unroll
        for (int kk = 0; kk < CI / 32; ++kk) {
            // B (32x16 f16): N = l15, K(j) = (j>>3)*16 + half*8 + (j&7)
            V16 bfrag;
            const v8h* brow = (const v8h*)&ztr[nt * 16 + l15][kk * 32];
            bfrag.h[0] = brow[half];
            bfrag.h[1] = brow[2 + half];
            acc = __builtin_amdgcn_wmma_f32_16x16x32_f16(
                false, afrag[kk].v, false, bfrag.v, (short)0, acc, false, false);
        }
#pragma unroll
        for (int r = 0; r < 8; ++r) {
            // C/D: row = half*8 + r, col = l15
            int ch = wv * 16 + half * 8 + r;
            int p  = p0 + nt * 16 + l15;
            float v = acc[r];
            y[((size_t)b * 64 + ch) * PP + p] = v;
            sacc[r] += v;
            qacc[r] += v * v;
        }
    }

    // One cross-lane reduction per owned channel (16 lanes share (half,r)).
#pragma unroll
    for (int r = 0; r < 8; ++r) {
        float s = sacc[r], q = qacc[r];
#pragma unroll
        for (int m = 8; m >= 1; m >>= 1) {
            s += __shfl_xor(s, m, 32);
            q += __shfl_xor(q, m, 32);
        }
        if (l15 == 0) {
            int ch = wv * 16 + half * 8 + r;
            atomicAdd(&ssum[ch], s);
            atomicAdd(&ssq[ch], q);
        }
    }
}

// ---------------- BN normalize (global stats) + SiLU -----------------------
__global__ void bnsilu_kernel(const float* __restrict__ y, const float* __restrict__ ssum,
                              const float* __restrict__ ssq, const float* __restrict__ g,
                              const float* __restrict__ bt, float* __restrict__ xo)
{
    int idx = blockIdx.x * blockDim.x + threadIdx.x;
    if (idx >= BB * 64 * PP) return;
    int ch = (idx >> 9) & 63;
    const float inv_n = 1.0f / (float)(BB * PP);
    float mu  = ssum[ch] * inv_n;
    float var = ssq[ch] * inv_n - mu * mu;
    float xn  = (y[idx] - mu) * rsqrtf(var + 1e-5f) * g[ch] + bt[ch];
    xo[idx] = xn / (1.0f + expf(-xn));
}

// --------- onset scan + weighted prototype scatter (one block per b) -------
__global__ void proto_kernel(const float* __restrict__ x, const int* __restrict__ flags,
                             float* __restrict__ out)
{
    __shared__ int   nid[PP];
    __shared__ int   rel[PP];
    __shared__ float wnt[NMAX];
    const int b   = blockIdx.x;
    const int tid = threadIdx.x;

    if (tid == 0) {
        int note = -1, last = -1;
        int len[NMAX];
        for (int n = 0; n < NMAX; ++n) len[n] = 0;
        for (int p = 0; p < PP; ++p) {
            if (flags[b * PP + p] == 1) { note++; last = p; }
            int r = (last >= 0) ? (p - last) : (p + 1);   // clip(p - last_on, 0, P-1)
            if (r > PP - 1) r = PP - 1;
            int id = (note >= 0 && note < NMAX) ? note : NMAX;
            nid[p] = id;
            rel[p] = r;
            if (id < NMAX) len[id]++;
        }
        float tot = 0.f;
        for (int n = 0; n < NMAX; ++n) tot += (float)len[n];
        for (int n = 0; n < NMAX; ++n) wnt[n] = (float)len[n] / (tot + 1e-7f);
    }
    __syncthreads();

    for (int idx = tid; idx < 64 * PP; idx += blockDim.x) {
        int c = idx >> 9, p = idx & 511;
        int id = nid[p];
        if (id < NMAX)
            atomicAdd(&out[((size_t)b * 64 + c) * PP + rel[p]],
                      x[((size_t)b * 64 + c) * PP + p] * wnt[id]);
    }
}

// ---------------------------------------------------------------------------
extern "C" void kernel_launch(void* const* d_in, const int* in_sizes, int n_in,
                              void* d_out, int out_size, void* d_ws, size_t ws_size,
                              hipStream_t stream)
{
    (void)in_sizes; (void)n_in; (void)ws_size;

    const float* wav = (const float*)d_in[0];
    const int*   onf = (const int*)d_in[1];
    const float* w0  = (const float*)d_in[2];
    const float* b0  = (const float*)d_in[3];
    const float* dw[4] = {(const float*)d_in[4],  (const float*)d_in[8],
                          (const float*)d_in[12], (const float*)d_in[16]};
    const float* pw[4] = {(const float*)d_in[5],  (const float*)d_in[9],
                          (const float*)d_in[13], (const float*)d_in[17]};
    const float* g[4]  = {(const float*)d_in[6],  (const float*)d_in[10],
                          (const float*)d_in[14], (const float*)d_in[18]};
    const float* bt[4] = {(const float*)d_in[7],  (const float*)d_in[11],
                          (const float*)d_in[15], (const float*)d_in[19]};

    float* ws    = (float*)d_ws;
    float* lr    = ws;                                  // 32*512
    float* x0    = lr  + (size_t)BB * PP;               // 32*32*512
    float* xa    = x0  + (size_t)BB * 32 * PP;          // 32*64*512
    float* xb    = xa  + (size_t)BB * 64 * PP;          // 32*64*512
    float* yb    = xb  + (size_t)BB * 64 * PP;          // 32*64*512
    float* stats = yb  + (size_t)BB * 64 * PP;          // 4 * 128

    hipMemsetAsync(stats, 0, 4 * 128 * sizeof(float), stream);
    hipMemsetAsync(d_out, 0, (size_t)out_size * sizeof(float), stream);

    rms_kernel<<<BB * PP / 8, 256, 0, stream>>>(wav, lr, BB * PP);
    conv0_kernel<<<(BB * 32 * PP + 255) / 256, 256, 0, stream>>>(lr, w0, b0, x0);

    const float* xin = x0;
    float* bufs[2] = {xa, xb};
    const int dils[4] = {1, 2, 4, 8};
    for (int n = 0; n < 4; ++n) {
        float* ss = stats + n * 128;
        float* sq = ss + 64;
        if (n == 0)
            dwpw_kernel<32><<<BB * 8, 128, 0, stream>>>(xin, dw[n], pw[n], yb, ss, sq, dils[n]);
        else
            dwpw_kernel<64><<<BB * 8, 128, 0, stream>>>(xin, dw[n], pw[n], yb, ss, sq, dils[n]);
        float* xo = bufs[n & 1];
        bnsilu_kernel<<<(BB * 64 * PP + 255) / 256, 256, 0, stream>>>(yb, ss, sq, g[n], bt[n], xo);
        xin = xo;
    }

    proto_kernel<<<BB, 256, 0, stream>>>(xin, onf, (float*)d_out);
}
